// HierarchicalLSTMAttDecoder_34617436406298
// MI455X (gfx1250) — matrix operand order repaired and checked by reading, not compiled, not run on hardware
//
#include <hip/hip_runtime.h>
#include <math.h>

// ---------------------------------------------------------------------------
// CDNA5 (gfx1250) wave32 WMMA types
// ---------------------------------------------------------------------------
typedef __attribute__((ext_vector_type(16))) __bf16 v16bf;
typedef __attribute__((ext_vector_type(8)))  float  v8f;

union FragAB { v16bf v; unsigned int u[8]; };

#define WPB 8    // waves per block (256 threads)
#define FC_NT 4  // N-tiles per wave in the vocab GEMM

__device__ __forceinline__ unsigned short f32_to_bf16(float f) {
    union { float f; unsigned int u; } x; x.f = f;
    unsigned int u = x.u;
    unsigned int r = u + 0x7FFFu + ((u >> 16) & 1u);   // round-to-nearest-even
    return (unsigned short)(r >> 16);
}
__device__ __forceinline__ float sigmoidf_(float x) { return 1.0f / (1.0f + __expf(-x)); }

// A-fragment K offset within a 32-wide K tile (16-bit A 16x32 layout, ISA 7.12.2)
__device__ __forceinline__ int a_koff(int v, int g) {
    return (v < 4) ? (2 * v + 8 * g) : (16 + 2 * (v - 4) + 8 * g);
}
// B-fragment K offset (same K order as the A fragment)
__device__ __forceinline__ int b_koff(int v, int g) { return a_koff(v, g); }

// ---------------------------------------------------------------------------
// Elementwise helpers
// ---------------------------------------------------------------------------
__global__ void k_f32_to_bf16(const float* __restrict__ src, unsigned short* __restrict__ dst, int n) {
    int i = blockIdx.x * blockDim.x + threadIdx.x;
    if (i < n) dst[i] = f32_to_bf16(src[i]);
}
__global__ void k_add_vec(const float* __restrict__ a, const float* __restrict__ b,
                          float* __restrict__ o, int n) {
    int i = blockIdx.x * blockDim.x + threadIdx.x;
    if (i < n) o[i] = a[i] + b[i];
}
__global__ void k_zero_f32(float* __restrict__ p, int n) {
    int i = blockIdx.x * blockDim.x + threadIdx.x;
    if (i < n) p[i] = 0.0f;
}
__global__ void k_zero_u16(unsigned short* __restrict__ p, int n) {
    int i = blockIdx.x * blockDim.x + threadIdx.x;
    if (i < n) p[i] = 0;
}

// LDS-tiled transpose+convert: src f32 [R,C] -> dst bf16 [C,R]
// Makes K-major weights N-major so GEMM B-fragments load as contiguous b128.
__global__ void k_transpose_bf16(const float* __restrict__ src, unsigned short* __restrict__ dst,
                                 int R, int C) {
    __shared__ unsigned short tile[32][33];          // +1 pad: no bank conflicts
    int rt = blockIdx.y * 32, ct = blockIdx.x * 32;
    int tx = threadIdx.x & 31, ty = threadIdx.x >> 5;  // 256 threads = 32x8
    for (int i = ty; i < 32; i += 8) {
        int r = rt + i, c = ct + tx;
        tile[i][tx] = (r < R && c < C) ? f32_to_bf16(src[(size_t)r * C + c]) : (unsigned short)0;
    }
    __syncthreads();
    for (int i = ty; i < 32; i += 8) {
        int r = ct + i, c = rt + tx;                 // dst is [C,R]
        if (r < C && c < R) dst[(size_t)r * R + c] = tile[tx][i];
    }
}

// Gather teacher-forcing word inputs: xemb[(s*12+w)*16+b][k]
//   w==0 -> emb[START=1]; else emb[reports[b][s][w-1]]
__global__ void k_gather_emb(const float* __restrict__ emb, const int* __restrict__ reports,
                             unsigned short* __restrict__ xemb) {
    int i = blockIdx.x * blockDim.x + threadIdx.x;
    if (i >= 5 * 12 * 16 * 1024) return;
    int k  = i & 1023;
    int b  = (i >> 10) & 15;
    int sw = i >> 14;                       // 0..59
    int s = sw / 12, w = sw % 12;
    int tok = (w == 0) ? 1 : reports[(b * 5 + s) * 12 + (w - 1)];
    xemb[i] = f32_to_bf16(emb[tok * 1024 + k]);
}

// ---------------------------------------------------------------------------
// Fused LSTM gate GEMM: gates[16,4096] = A1@W1^T + A2@W2^T + bias
//   A1,A2 : [16,1024] bf16 ; W1,W2 : [4096,1024] bf16 (row-major NxK)
// One wave per 16x16 N-tile, 64 WMMAs per wave.
// ---------------------------------------------------------------------------
__global__ void k_gemm_gates(const unsigned short* __restrict__ A1,
                             const unsigned short* __restrict__ A2,
                             const unsigned short* __restrict__ W1,
                             const unsigned short* __restrict__ W2,
                             const float* __restrict__ bias,
                             float* __restrict__ out) {
    const int K = 1024, N = 4096;
    int ntile = blockIdx.x * WPB + (threadIdx.x >> 5);
    int lane  = threadIdx.x & 31;
    int g = lane >> 4, m = lane & 15;
    int n = ntile * 16 + m;

    v8f acc = {};
    const unsigned short* Aps[2] = { A1, A2 };
    const unsigned short* Wps[2] = { W1, W2 };
    #pragma unroll
    for (int p = 0; p < 2; ++p) {
        const unsigned short* Arow = Aps[p] + m * K;
        const unsigned short* Wrow = Wps[p] + n * K;
        for (int kb = 0; kb < K; kb += 32) {
            FragAB a, bm;
            #pragma unroll
            for (int v = 0; v < 8; ++v) {
                a.u[v]  = *(const unsigned int*)(Arow + kb + a_koff(v, g));
                bm.u[v] = *(const unsigned int*)(Wrow + kb + b_koff(v, g));
            }
            acc = __builtin_amdgcn_wmma_f32_16x16x32_bf16(
                false, a.v, false, bm.v, (short)0, acc, false, false);
        }
    }
    float bn = bias[n];
    #pragma unroll
    for (int r = 0; r < 8; ++r) {
        int mm = r + 8 * g;                 // batch row
        out[mm * N + n] = acc[r] + bn;
    }
}

// LSTM pointwise: PyTorch gate order i,f,g,o
__global__ void k_lstm_cell(const float* __restrict__ gates,   // [16,4096]
                            const float* __restrict__ c_in,    // [16,1024]
                            float* __restrict__ c_out,
                            float* __restrict__ h_f32,         // nullable
                            unsigned short* __restrict__ h_bf16) {
    int i = blockIdx.x * blockDim.x + threadIdx.x;
    if (i >= 16 * 1024) return;
    int b = i >> 10, j = i & 1023;
    const float* gr = gates + b * 4096;
    float gi = gr[j], gf = gr[j + 1024], gg = gr[j + 2048], go = gr[j + 3072];
    float c = sigmoidf_(gf) * c_in[i] + sigmoidf_(gi) * tanhf(gg);
    float h = sigmoidf_(go) * tanhf(c);
    c_out[i] = c;
    if (h_f32) h_f32[i] = h;
    h_bf16[i] = f32_to_bf16(h);
}

// ---------------------------------------------------------------------------
// Attention layer 1: t1[784,1024] = tanh(concat(features,h) @ w1 + b1)
//   feat [784,1024] bf16, hbf [16,1024] bf16, W1t [1024,2048] bf16 (N-major)
// Row r = b*49+nl; concat handled by two K segments (no xcat materialization).
// ---------------------------------------------------------------------------
__global__ void k_attn_gemm1(const unsigned short* __restrict__ feat,
                             const unsigned short* __restrict__ hbf,
                             const unsigned short* __restrict__ W1t,  // [1024,2048]
                             const float* __restrict__ b1,
                             float* __restrict__ t1) {
    const int K = 1024, N = 1024;
    int wid   = blockIdx.x * WPB + (threadIdx.x >> 5);
    int mtile = wid / 64;                   // 0..48
    int ntile = wid % 64;                   // 0..63
    int lane  = threadIdx.x & 31;
    int g = lane >> 4, m = lane & 15;
    int row  = mtile * 16 + m;              // 0..783
    int bidx = row / 49;
    int n    = ntile * 16 + m;
    const unsigned short* Wrow = W1t + (size_t)n * 2048;

    v8f acc = {};
    #pragma unroll
    for (int seg = 0; seg < 2; ++seg) {
        const unsigned short* Arow = seg ? (hbf + bidx * 1024) : (feat + row * 1024);
        int kofs = seg * 1024;              // offset into w1's K dim
        for (int kb = 0; kb < K; kb += 32) {
            FragAB a, bm;
            #pragma unroll
            for (int v = 0; v < 8; ++v) {
                a.u[v]  = *(const unsigned int*)(Arow + kb + a_koff(v, g));
                bm.u[v] = *(const unsigned int*)(Wrow + kofs + kb + b_koff(v, g));
            }
            acc = __builtin_amdgcn_wmma_f32_16x16x32_bf16(
                false, a.v, false, bm.v, (short)0, acc, false, false);
        }
    }
    float bn = b1[n];
    #pragma unroll
    for (int r = 0; r < 8; ++r) {
        int rr = mtile * 16 + r + 8 * g;
        t1[rr * N + n] = tanhf(acc[r] + bn);
    }
}

// Attention finish (1 block per batch b): e = t1@w2+b2, softmax over 49,
// write scores (if s>=0), att_bf16 = scores @ features
__global__ void k_attn_finish(const float* __restrict__ t1,     // [784,1024]
                              const float* __restrict__ w2,     // [1024]
                              const float* __restrict__ b2,     // [1]
                              const float* __restrict__ feat32, // [16,49,1024] f32
                              float* __restrict__ scores_out,   // [B,S,49]
                              unsigned short* __restrict__ att_bf, // [16,1024]
                              int s) {
    __shared__ float e[49];
    __shared__ float sm[49];
    __shared__ float red[256];
    int b = blockIdx.x, t = threadIdx.x;
    for (int nl = 0; nl < 49; ++nl) {
        const float* row = t1 + (b * 49 + nl) * 1024;
        float p = 0.f;
        for (int k = t; k < 1024; k += 256) p += row[k] * w2[k];
        red[t] = p; __syncthreads();
        for (int off = 128; off > 0; off >>= 1) {
            if (t < off) red[t] += red[t + off];
            __syncthreads();
        }
        if (t == 0) e[nl] = red[0] + b2[0];
        __syncthreads();
    }
    if (t == 0) {
        float mx = e[0];
        for (int i = 1; i < 49; ++i) mx = fmaxf(mx, e[i]);
        float sum = 0.f;
        for (int i = 0; i < 49; ++i) { sm[i] = __expf(e[i] - mx); sum += sm[i]; }
        float inv = 1.0f / sum;
        for (int i = 0; i < 49; ++i) sm[i] *= inv;
    }
    __syncthreads();
    if (s >= 0)
        for (int nl = t; nl < 49; nl += 256) scores_out[(b * 5 + s) * 49 + nl] = sm[nl];
    for (int f = t; f < 1024; f += 256) {
        float acc = 0.f;
        for (int nl = 0; nl < 49; ++nl) acc += sm[nl] * feat32[(b * 49 + nl) * 1024 + f];
        att_bf[b * 1024 + f] = f32_to_bf16(acc);
    }
}

// stop[b][s] = sigmoid(h_sent[b] . stop_w + stop_b)
__global__ void k_stop(const float* __restrict__ h, const float* __restrict__ stop_w,
                       const float* __restrict__ stop_b, float* __restrict__ stops_out, int s) {
    __shared__ float red[256];
    int b = blockIdx.x, t = threadIdx.x;
    float p = 0.f;
    for (int k = t; k < 1024; k += 256) p += h[b * 1024 + k] * stop_w[k];
    red[t] = p; __syncthreads();
    for (int off = 128; off > 0; off >>= 1) {
        if (t < off) red[t] += red[t + off];
        __syncthreads();
    }
    if (t == 0) stops_out[b * 5 + s] = sigmoidf_(red[0] + stop_b[0]);
}

// ---------------------------------------------------------------------------
// Final vocab projection (all 960 hidden states at once):
//   seq[b][s][w][v] = Hall[(s*12+w)*16+b] @ fc_w + fc_b
// Wt is the transposed weight [15000,1024] bf16 (N-major -> contiguous b128).
// Each wave: 4 N-tiles sharing one A fragment (4 wmma per A load).
// ---------------------------------------------------------------------------
__global__ void k_fc_gemm(const unsigned short* __restrict__ Hall, // [960,1024] bf16
                          const unsigned short* __restrict__ Wt,   // [15000,1024] bf16
                          const float* __restrict__ fc_b,          // [15000]
                          float* __restrict__ out) {               // seq region of d_out
    const int K = 1024, V = 15000;
    const int NG = 235;                     // ceil(938 n-tiles / FC_NT)
    int wid = blockIdx.x * WPB + (threadIdx.x >> 5);
    if (wid >= 60 * NG) return;             // wave-uniform guard
    int mtile = wid / NG;                   // 0..59  (s*12+w)
    int ngrp  = wid % NG;
    int lane = threadIdx.x & 31;
    int g = lane >> 4, m = lane & 15;

    int ncol[FC_NT];
    const unsigned short* Wrow[FC_NT];
    v8f acc[FC_NT] = {};
    #pragma unroll
    for (int t = 0; t < FC_NT; ++t) {
        int n = (ngrp * FC_NT + t) * 16 + m;
        ncol[t] = n;
        int nc = (n < V) ? n : (V - 1);     // clamp loads; stores are guarded
        Wrow[t] = Wt + (size_t)nc * K;
    }
    const unsigned short* Arow = Hall + (size_t)(mtile * 16 + m) * K;
    for (int kb = 0; kb < K; kb += 32) {
        FragAB a;
        #pragma unroll
        for (int v = 0; v < 8; ++v)
            a.u[v] = *(const unsigned int*)(Arow + kb + a_koff(v, g));
        #pragma unroll
        for (int t = 0; t < FC_NT; ++t) {
            FragAB bm;
            #pragma unroll
            for (int v = 0; v < 8; ++v)
                bm.u[v] = *(const unsigned int*)(Wrow[t] + kb + b_koff(v, g));
            acc[t] = __builtin_amdgcn_wmma_f32_16x16x32_bf16(
                false, a.v, false, bm.v, (short)0, acc[t], false, false);
        }
    }
    #pragma unroll
    for (int t = 0; t < FC_NT; ++t) {
        int n = ncol[t];
        if (n >= V) continue;
        float bn = fc_b[n];
        #pragma unroll
        for (int r = 0; r < 8; ++r) {
            int bb = r + 8 * g;             // batch
            out[((size_t)bb * 60 + mtile) * V + n] = acc[t][r] + bn;
        }
    }
}

// ---------------------------------------------------------------------------
// Host orchestration
// ---------------------------------------------------------------------------
static inline int cdiv(int a, int b) { return (a + b - 1) / b; }

extern "C" void kernel_launch(void* const* d_in, const int* in_sizes, int n_in,
                              void* d_out, int out_size, void* d_ws, size_t ws_size,
                              hipStream_t stream) {
    const float* features = (const float*)d_in[0];   // [16,49,1024]
    const int*   reports  = (const int*)  d_in[1];   // [16,5,12]
    const float* attn_w1  = (const float*)d_in[2];   // [2048,1024]
    const float* attn_b1  = (const float*)d_in[3];   // [1024]
    const float* attn_w2  = (const float*)d_in[4];   // [1024,1]
    const float* attn_b2  = (const float*)d_in[5];   // [1]
    const float* s_wih    = (const float*)d_in[6];   // [4096,1024]
    const float* s_whh    = (const float*)d_in[7];   // [4096,1024]
    const float* s_bih    = (const float*)d_in[8];   // [4096]
    const float* s_bhh    = (const float*)d_in[9];   // [4096]
    const float* stop_w   = (const float*)d_in[10];  // [1024,1]
    const float* stop_b   = (const float*)d_in[11];  // [1]
    const float* emb      = (const float*)d_in[12];  // [15000,1024]
    const float* w_wih    = (const float*)d_in[13];  // [4096,1024]
    const float* w_whh    = (const float*)d_in[14];  // [4096,1024]
    const float* w_bih    = (const float*)d_in[15];  // [4096]
    const float* w_bhh    = (const float*)d_in[16];  // [4096]
    const float* fc_w     = (const float*)d_in[17];  // [1024,15000]
    const float* fc_b     = (const float*)d_in[18];  // [15000]

    float* out        = (float*)d_out;
    float* out_seq    = out;                         // 16*5*12*15000
    float* out_stops  = out + 14400000;              // 16*5
    float* out_scores = out + 14400080;              // 16*5*49

    // ---- workspace layout ----
    char* ws = (char*)d_ws;
    size_t off = 0;
    auto alloc = [&](size_t bytes) { size_t o = off; off = (off + bytes + 255) & ~(size_t)255; return o; };
    unsigned short* fcwt_bf = (unsigned short*)(ws + alloc((size_t)15000 * 1024 * 2)); // transposed
    unsigned short* wwih_bf = (unsigned short*)(ws + alloc((size_t)4096 * 1024 * 2));
    unsigned short* wwhh_bf = (unsigned short*)(ws + alloc((size_t)4096 * 1024 * 2));
    unsigned short* swih_bf = (unsigned short*)(ws + alloc((size_t)4096 * 1024 * 2));
    unsigned short* swhh_bf = (unsigned short*)(ws + alloc((size_t)4096 * 1024 * 2));
    unsigned short* aw1t_bf = (unsigned short*)(ws + alloc((size_t)1024 * 2048 * 2));  // transposed
    unsigned short* feat_bf = (unsigned short*)(ws + alloc((size_t)784 * 1024 * 2));
    unsigned short* xemb    = (unsigned short*)(ws + alloc((size_t)60 * 16 * 1024 * 2));
    unsigned short* Hall    = (unsigned short*)(ws + alloc((size_t)960 * 1024 * 2));
    float* t1    = (float*)(ws + alloc((size_t)784 * 1024 * 4));
    float* gates = (float*)(ws + alloc((size_t)16 * 4096 * 4));
    float* h_s   = (float*)(ws + alloc((size_t)16 * 1024 * 4));
    float* c_s   = (float*)(ws + alloc((size_t)16 * 1024 * 4));
    unsigned short* hs_bf = (unsigned short*)(ws + alloc((size_t)16 * 1024 * 2));
    float* c_w   = (float*)(ws + alloc((size_t)16 * 1024 * 4));
    unsigned short* att_bf = (unsigned short*)(ws + alloc((size_t)16 * 1024 * 2));
    float* sb    = (float*)(ws + alloc((size_t)4096 * 4));
    float* wb    = (float*)(ws + alloc((size_t)4096 * 4));
    (void)ws_size; (void)in_sizes; (void)n_in; (void)out_size;

    // ---- setup: weight conversion (+transpose for K-major weights), bias
    //      fusion, embedding gather, zero state
    auto cvt = [&](const float* s, unsigned short* d, int n) {
        k_f32_to_bf16<<<cdiv(n, 256), 256, 0, stream>>>(s, d, n);
    };
    cvt(w_wih,   wwih_bf, 4096 * 1024);
    cvt(w_whh,   wwhh_bf, 4096 * 1024);
    cvt(s_wih,   swih_bf, 4096 * 1024);
    cvt(s_whh,   swhh_bf, 4096 * 1024);
    cvt(features, feat_bf, 784 * 1024);
    // fc_w [1024,15000] -> [15000,1024]; attn_w1 [2048,1024] -> [1024,2048]
    k_transpose_bf16<<<dim3(cdiv(15000, 32), cdiv(1024, 32)), 256, 0, stream>>>(
        fc_w, fcwt_bf, 1024, 15000);
    k_transpose_bf16<<<dim3(cdiv(1024, 32), cdiv(2048, 32)), 256, 0, stream>>>(
        attn_w1, aw1t_bf, 2048, 1024);
    k_add_vec<<<16, 256, 0, stream>>>(s_bih, s_bhh, sb, 4096);
    k_add_vec<<<16, 256, 0, stream>>>(w_bih, w_bhh, wb, 4096);
    k_gather_emb<<<cdiv(60 * 16 * 1024, 256), 256, 0, stream>>>(emb, reports, xemb);
    k_zero_f32<<<64, 256, 0, stream>>>(h_s, 16 * 1024);
    k_zero_f32<<<64, 256, 0, stream>>>(c_s, 16 * 1024);
    k_zero_u16<<<64, 256, 0, stream>>>(hs_bf, 16 * 1024);

    // ---- att0 = attention(features, h=0)  (scores discarded: s=-1)
    k_attn_gemm1<<<392, 256, 0, stream>>>(feat_bf, hs_bf, aw1t_bf, attn_b1, t1);
    k_attn_finish<<<16, 256, 0, stream>>>(t1, attn_w2, attn_b2, features,
                                          out_scores, att_bf, -1);

    // ---- sequential chain: 5 sentences x (sent-LSTM + 12 word-LSTM steps)
    for (int s = 0; s < 5; ++s) {
        // sentence LSTM: gates = att@s_wih^T + h@s_whh^T + (bih+bhh)
        k_gemm_gates<<<32, 256, 0, stream>>>(att_bf, hs_bf, swih_bf, swhh_bf, sb, gates);
        k_lstm_cell<<<64, 256, 0, stream>>>(gates, c_s, c_s, h_s, hs_bf);
        // word LSTM: h0 = sentence h (topic), c0 = 0, input0 = <start> emb
        k_zero_f32<<<64, 256, 0, stream>>>(c_w, 16 * 1024);
        for (int w = 0; w < 12; ++w) {
            const unsigned short* xin = xemb + (size_t)(s * 12 + w) * 16 * 1024;
            const unsigned short* hin = (w == 0) ? hs_bf
                                                 : Hall + (size_t)(s * 12 + w - 1) * 16 * 1024;
            unsigned short* hout = Hall + (size_t)(s * 12 + w) * 16 * 1024;
            k_gemm_gates<<<32, 256, 0, stream>>>(xin, hin, wwih_bf, wwhh_bf, wb, gates);
            k_lstm_cell<<<64, 256, 0, stream>>>(gates, c_w, c_w, nullptr, hout);
        }
        // stop gate on post-LSTM sentence h
        k_stop<<<16, 256, 0, stream>>>(h_s, stop_w, stop_b, out_stops, s);
        // next attention (its scores are this sentence's score output)
        k_attn_gemm1<<<392, 256, 0, stream>>>(feat_bf, hs_bf, aw1t_bf, attn_b1, t1);
        k_attn_finish<<<16, 256, 0, stream>>>(t1, attn_w2, attn_b2, features,
                                              out_scores, att_bf, s);
    }

    // ---- one massive deferred vocab projection: [960,1024] @ [1024,15000]
    // 60 M-tiles x 235 n-groups (4 tiles each) = 14100 waves / 8 per block
    k_fc_gemm<<<cdiv(60 * 235, WPB), 256, 0, stream>>>(Hall, fcwt_bf, fc_b, out_seq);
}
